// SelfAttentionLayer_23983097381577
// MI455X (gfx1250) — matrix operand order repaired and checked by reading, not compiled, not run on hardware
//
#include <hip/hip_runtime.h>
#include <stddef.h>

// ---------------- problem constants ----------------
#define B_DIM 32
#define H_DIM 64
#define W_DIM 64
#define C_DIM 256

// LDS pitches (in elements): stagger banks, keep 16B alignment
#define XP  264   // Xb / Qb / Kb row pitch (bf16)
#define VP  72    // V-transposed row pitch (bf16)
#define APF 68    // attn f32 pitch
#define APB 72    // attn bf16 pitch

typedef __bf16          v16bf  __attribute__((ext_vector_type(16)));
typedef float           v8f    __attribute__((ext_vector_type(8)));
typedef float           f32x4  __attribute__((ext_vector_type(4)));
typedef unsigned short  u16x4  __attribute__((ext_vector_type(4)));
typedef unsigned short  u16x8  __attribute__((ext_vector_type(8)));
typedef unsigned short  u16x16 __attribute__((ext_vector_type(16)));

// native f32 -> bf16 (RNE): let the backend pick the hardware cvt op
__device__ __forceinline__ unsigned short f2bf(float f) {
  __bf16 h = (__bf16)f;
  return __builtin_bit_cast(unsigned short, h);
}

// A-matrix fragment (16x32 bf16, MxK): element i <-> K = 16*half + i
__device__ __forceinline__ v16bf frag_a(const unsigned short* row, int k0, int half) {
  u16x8 lo = *(const u16x8*)(row + k0 + 16 * half);
  u16x8 hi = *(const u16x8*)(row + k0 + 16 * half + 8);
  u16x16 u;
#pragma unroll
  for (int i = 0; i < 8; ++i) { u[i] = lo[i]; u[8 + i] = hi[i]; }
  return __builtin_bit_cast(v16bf, u);
}

// B-matrix fragment (32x16 bf16, KxN): lane holds row N, element i <-> K = 16*half + i
__device__ __forceinline__ v16bf frag_b(const unsigned short* row, int k0, int half) {
  u16x8 lo = *(const u16x8*)(row + k0 + 16 * half);
  u16x8 hi = *(const u16x8*)(row + k0 + 16 * half + 8);
  u16x16 u;
#pragma unroll
  for (int i = 0; i < 8; ++i) { u[i] = lo[i]; u[8 + i] = hi[i]; }
  return __builtin_bit_cast(v16bf, u);
}

// -------- pre-pass: fp32 weights -> bf16 B-fragment layout in d_ws --------
// layout: [p(3)][kk(8)][ntile(16)][lane(32)][elem(16)]
__global__ void sa_pack_weights(const float* __restrict__ wq,
                                const float* __restrict__ wk,
                                const float* __restrict__ wv,
                                unsigned short* __restrict__ wpack) {
  int gid = blockIdx.x * blockDim.x + threadIdx.x;
  if (gid >= 3 * 8 * 16 * 32) return;
  int lane  = gid & 31;
  int ntile = (gid >> 5) & 15;
  int kk    = (gid >> 9) & 7;
  int p     = gid >> 12;
  const float* w = (p == 0) ? wq : (p == 1) ? wk : wv;
  int half = lane >> 4, n = lane & 15;
  int N = ntile * 16 + n;
  unsigned short* dst = wpack + (size_t)gid * 16;
#pragma unroll
  for (int i = 0; i < 16; ++i) {
    int K = kk * 32 + 16 * half + i;
    dst[i] = f2bf(w[K * C_DIM + N]);
  }
}

// ---------------- main fused kernel: one workgroup per (b,h) ----------------
// LDS layout (138,240 B total -> 2 workgroups per 320 KB WGP):
//   region0 [33792 B]: Xb[64][XP]   (stages 0-1)   OVERLAID WITH
//                      attnF[64][APF] + attnB[64][APB] + red[64][4] (stages 2+)
//   Qb[64][XP], Kb[64][XP], Vt[256][VP]
__global__ void sa_fused_attn(const float* __restrict__ x,
                              const float* __restrict__ bq,
                              const float* __restrict__ bk,
                              const float* __restrict__ bv,
                              const unsigned short* __restrict__ wpack,
                              float* __restrict__ out) {
  extern __shared__ unsigned char smem[];
  unsigned short* Xb    = (unsigned short*)smem;                     // [64][XP]
  unsigned short* Qb    = (unsigned short*)(smem + 33792);           // [64][XP]
  unsigned short* Kb    = (unsigned short*)(smem + 2 * 33792);       // [64][XP]
  unsigned short* Vt    = (unsigned short*)(smem + 3 * 33792);       // [256][VP]
  // overlay of region0 (Xb dead after stage 1):
  float*          attnF = (float*)smem;                              // [64][APF] = 17408 B
  unsigned short* attnB = (unsigned short*)(smem + 17408);           // [64][APB] =  9216 B
  float*          red   = (float*)(smem + 17408 + 9216);             // [64][4]   =  1024 B

  const int tid  = threadIdx.x;
  const int lane = tid & 31;
  const int wave = __builtin_amdgcn_readfirstlane(tid >> 5);  // scalar wave id
  const int half = lane >> 4;
  const int ln   = lane & 15;

  const float* xg = x   + (size_t)blockIdx.x * W_DIM * C_DIM;
  float*       og = out + (size_t)blockIdx.x * W_DIM * C_DIM;

  // ---- load X slice, convert to bf16 in LDS ----
  for (int i = tid; i < W_DIM * (C_DIM / 4); i += 256) {
    int row = i >> 6;
    int c4  = (i & 63) << 2;
    f32x4 v = *(const f32x4*)(xg + row * C_DIM + c4);
    u16x4 s;
    s[0] = f2bf(v[0]); s[1] = f2bf(v[1]); s[2] = f2bf(v[2]); s[3] = f2bf(v[3]);
    *(u16x4*)(Xb + row * XP + c4) = s;
  }
  __syncthreads();

  // ---- stage 1: Q/K/V projections. Each wave owns (p, ntile) column-jobs and
  //      keeps the 8 K-step weight B-fragments resident in VGPRs across 4 M-tiles.
  for (int job = wave; job < 48; job += 8) {
    int p     = job >> 4;          // 0=Q, 1=K, 2=V   (scalar)
    int ntile = job & 15;
    int N     = ntile * 16 + ln;

    const float* bias = (p == 0) ? bq : (p == 1) ? bk : bv;
    float b0 = bias[N];

    // resident weight fragments: 8 x (16 bf16) = 64 VGPRs
    v16bf bfr[8];
    const unsigned short* wp = wpack + ((size_t)((p * 8) * 16 + ntile) * 32 + lane) * 16;
#pragma unroll
    for (int kk = 0; kk < 8; ++kk)
      bfr[kk] = __builtin_bit_cast(v16bf, *(const u16x16*)(wp + (size_t)kk * (16 * 32 * 16)));

#pragma unroll
    for (int mtile = 0; mtile < 4; ++mtile) {
      v8f acc;
#pragma unroll
      for (int r = 0; r < 8; ++r) acc[r] = b0;
      const unsigned short* arow = Xb + (mtile * 16 + ln) * XP;
#pragma unroll
      for (int kk = 0; kk < 8; ++kk) {
        v16bf a = frag_a(arow, kk * 32, half);
        acc = __builtin_amdgcn_wmma_f32_16x16x32_bf16(false, a, false, bfr[kk], (short)0, acc, false, false);
      }
      if (p == 2) { // V: store transposed, one contiguous 16B ds store per lane
        u16x8 s;
#pragma unroll
        for (int r = 0; r < 8; ++r) s[r] = f2bf(acc[r]);
        *(u16x8*)(Vt + N * VP + mtile * 16 + 8 * half) = s;
      } else {      // Q/K: row-major bf16
        unsigned short* dst = (p == 0) ? Qb : Kb;
#pragma unroll
        for (int r = 0; r < 8; ++r)
          dst[(mtile * 16 + 8 * half + r) * XP + N] = f2bf(acc[r]);
      }
    }
  }
  __syncthreads();   // Xb dead from here; region0 becomes attnF/attnB/red

  // ---- stage 2: attn[w,v] = sum_c K[w,c] * Q[v,c]  (16 tiles over 8 waves) ----
#pragma unroll
  for (int tt = 0; tt < 2; ++tt) {
    int t2    = wave * 2 + tt;
    int mtile = t2 >> 2;
    int ntile = t2 & 3;
    v8f acc;
#pragma unroll
    for (int r = 0; r < 8; ++r) acc[r] = 0.0f;
    const unsigned short* arow = Kb + (mtile * 16 + ln) * XP;
    const unsigned short* brow = Qb + (ntile * 16 + ln) * XP;
#pragma unroll
    for (int kk = 0; kk < 8; ++kk) {
      v16bf a = frag_a(arow, kk * 32, half);
      v16bf b = frag_b(brow, kk * 32, half);
      acc = __builtin_amdgcn_wmma_f32_16x16x32_bf16(false, a, false, b, (short)0, acc, false, false);
    }
#pragma unroll
    for (int r = 0; r < 8; ++r)
      attnF[(mtile * 16 + 8 * half + r) * APF + ntile * 16 + ln] = acc[r];
  }
  __syncthreads();

  // ---- softmax over v (row-wise), 4 threads per row ----
  {
    int r = tid >> 2, part = tid & 3;
    float* rowp = attnF + r * APF + part * 16;
    float m = -3.402823466e38f;
#pragma unroll
    for (int i = 0; i < 16; ++i) m = fmaxf(m, rowp[i]);
    red[r * 4 + part] = m;
    __syncthreads();
    float mrow = fmaxf(fmaxf(red[r * 4 + 0], red[r * 4 + 1]),
                       fmaxf(red[r * 4 + 2], red[r * 4 + 3]));
    float s = 0.0f;
#pragma unroll
    for (int i = 0; i < 16; ++i) {
      float e = __expf(rowp[i] - mrow);
      rowp[i] = e;
      s += e;
    }
    __syncthreads();                 // all max reads done before red reuse
    red[r * 4 + part] = s;
    __syncthreads();
    float inv = 1.0f / (red[r * 4 + 0] + red[r * 4 + 1] + red[r * 4 + 2] + red[r * 4 + 3]);
    unsigned short* brow = attnB + r * APB + part * 16;
#pragma unroll
    for (int i = 0; i < 16; ++i) brow[i] = f2bf(rowp[i] * inv);
  }
  __syncthreads();

  // ---- stage 3: out[w,c] = (sum_v attn[w,v] * V[v,c]) * x[w,c]  (64 tiles over 8 waves) ----
  for (int j = 0; j < 8; ++j) {
    int t3    = wave + (j << 3);
    int mtile = t3 >> 4;
    int ntile = t3 & 15;
    v8f acc;
#pragma unroll
    for (int r = 0; r < 8; ++r) acc[r] = 0.0f;
    const unsigned short* arow = attnB + (mtile * 16 + ln) * APB;
    const unsigned short* brow = Vt + (ntile * 16 + ln) * VP;   // Vt row = channel c, contiguous in v
#pragma unroll
    for (int kk = 0; kk < 2; ++kk) {
      v16bf a = frag_a(arow, kk * 32, half);
      v16bf b = frag_b(brow, kk * 32, half);
      acc = __builtin_amdgcn_wmma_f32_16x16x32_bf16(false, a, false, b, (short)0, acc, false, false);
    }
    int ccol = ntile * 16 + ln;
#pragma unroll
    for (int r = 0; r < 8; ++r) {
      int w = mtile * 16 + 8 * half + r;
      float xv = xg[w * C_DIM + ccol];           // fp32 gate with raw input
      og[w * C_DIM + ccol] = acc[r] * xv;
    }
  }
}

// ---------------- host-side launch ----------------
extern "C" void kernel_launch(void* const* d_in, const int* in_sizes, int n_in,
                              void* d_out, int out_size, void* d_ws, size_t ws_size,
                              hipStream_t stream) {
  (void)in_sizes; (void)n_in; (void)out_size; (void)ws_size;
  const float* x  = (const float*)d_in[0];
  const float* wq = (const float*)d_in[1];
  const float* bq = (const float*)d_in[2];
  const float* wk = (const float*)d_in[3];
  const float* bk = (const float*)d_in[4];
  const float* wv = (const float*)d_in[5];
  const float* bv = (const float*)d_in[6];
  unsigned short* wpack = (unsigned short*)d_ws;  // 3 * 256*256 bf16 = 384 KB

  sa_pack_weights<<<48, 256, 0, stream>>>(wq, wk, wv, wpack);

  const size_t lds_bytes =
      (size_t)33792 +                  // region0: Xb  (>= attnF+attnB+red = 27648)
      (size_t)33792 +                  // Qb
      (size_t)33792 +                  // Kb
      (size_t)(C_DIM * VP) * 2;        // Vt = 36864   -> total 138,240 B
  sa_fused_attn<<<B_DIM * H_DIM, 256, lds_bytes, stream>>>(x, bq, bk, bv, wpack, (float*)d_out);
}
